// ChunkedLinearDeltaMemory_56513179680989
// MI455X (gfx1250) — matrix operand order repaired and checked by reading, not compiled, not run on hardware
//
#include <hip/hip_runtime.h>
#include <math.h>

// ---------------- problem constants ----------------
#define DDIM     512
#define CHUNK_T  64      // tokens per chunk
#define NCHUNK   64
#define NBATCH   16
#define RBLK     8       // workgroups (row blocks) per batch
#define ROWS     64      // d_out rows per workgroup
#define MAX_LR   0.2f
#define MIN_DEC  0.5f
#define MAX_NRM  30.0f

#define MF_PITCH 520     // fp32 M row pitch in dwords (512 + 2x4 TDM pad)
#define BF_PITCH 257     // bf16 row pitch in dwords (514 halves)

typedef __attribute__((ext_vector_type(16))) __bf16        v16bf;
typedef __attribute__((ext_vector_type(8)))  float         v8f;
typedef __attribute__((ext_vector_type(4)))  unsigned int  v4u;
typedef __attribute__((ext_vector_type(8)))  int           v8i_t;
typedef __attribute__((ext_vector_type(4)))  int           v4i_t;

#if __has_builtin(__builtin_amdgcn_tensor_load_to_lds) && __has_builtin(__builtin_amdgcn_tensor_store_from_lds)
#define USE_TDM 1
#else
#define USE_TDM 0
#endif

#if __has_builtin(__builtin_amdgcn_s_wait_tensorcnt)
#define WAIT_TENSOR() __builtin_amdgcn_s_wait_tensorcnt(0)
#else
#define WAIT_TENSOR() ((void)0)
#endif

__device__ __forceinline__ float sigm(float z) { return 1.0f / (1.0f + __expf(-z)); }

__device__ __forceinline__ unsigned pk_bf16(float a, float b) {
    union { unsigned u; __bf16 h[2]; } p;
    p.h[0] = (__bf16)a; p.h[1] = (__bf16)b;
    return p.u;
}
__device__ __forceinline__ float bf_lo(unsigned u) {
    union { unsigned u; __bf16 h[2]; } p; p.u = u; return (float)p.h[0];
}
__device__ __forceinline__ float bf_hi(unsigned u) {
    union { unsigned u; __bf16 h[2]; } p; p.u = u; return (float)p.h[1];
}
__device__ __forceinline__ unsigned lds_off(const void* p) {
    // generic "shared" pointer: low 32 bits are the LDS offset (aperture mapping)
    return (unsigned)(unsigned long long)(size_t)p;
}

#if USE_TDM
// Pack TDM D# groups 0/1 per CDNA5 ISA 8.3/8.4 (2-D tile, data_size=4B).
__device__ __forceinline__ void tdm_desc(unsigned lds_addr, unsigned long long gaddr,
                                         unsigned td0, unsigned td1,
                                         unsigned tile0, unsigned tile1,
                                         unsigned long long stride0,
                                         int pad_en, int pad_interval, int pad_amount,
                                         v4u& g0, v8i_t& g1)
{
    g0.x = 1u;                                                    // count = 1
    g0.y = lds_addr;
    g0.z = (unsigned)(gaddr & 0xFFFFFFFFull);
    g0.w = (unsigned)((gaddr >> 32) & 0x1FFFFFFull) | (2u << 30); // type = 2 (image)

    unsigned long long q0 = ((unsigned long long)2u << 16)                      // data_size = 4B
                          | ((unsigned long long)(pad_en & 1) << 20)
                          | ((unsigned long long)(pad_interval & 7) << 22)
                          | ((unsigned long long)(pad_amount & 127) << 25)
                          | ((unsigned long long)(td0 & 0xFFFFu) << 48);        // tensor_dim0[15:0]
    unsigned long long q1 = (unsigned long long)((td0 >> 16) & 0xFFFFu)         // tensor_dim0[31:16]
                          | ((unsigned long long)td1 << 16)                     // tensor_dim1
                          | ((unsigned long long)(tile0 & 0xFFFFu) << 48);      // tile_dim0
    unsigned long long q2 = (unsigned long long)(tile1 & 0xFFFFu)               // tile_dim1 (tile_dim2=0)
                          | ((stride0 & 0xFFFFFFFFull) << 32);                  // dim0_stride[31:0]
    unsigned long long q3 = ((stride0 >> 32) & 0xFFFFull);                      // dim0_stride[47:32]
    g1[0] = (int)(unsigned)q0; g1[1] = (int)(unsigned)(q0 >> 32);
    g1[2] = (int)(unsigned)q1; g1[3] = (int)(unsigned)(q1 >> 32);
    g1[4] = (int)(unsigned)q2; g1[5] = (int)(unsigned)(q2 >> 32);
    g1[6] = (int)(unsigned)q3; g1[7] = (int)(unsigned)(q3 >> 32);
}
#endif

extern "C" __global__ __launch_bounds__(256)
void delta_mem_kernel(const float* __restrict__ x,
                      const float* __restrict__ Minit,
                      const float* __restrict__ eta_w,  const float* __restrict__ eta_bp,
                      const float* __restrict__ alpha_w,const float* __restrict__ alpha_bp,
                      const float* __restrict__ gate_w, const float* __restrict__ gate_bp,
                      float* __restrict__ out,          // [16,4096,512]
                      float* __restrict__ Mfin,         // [16,512,512]
                      int*   __restrict__ arrive,       // [16*64]
                      float* __restrict__ partials)     // [16*64*8]
{
    extern __shared__ char smem[];
    float*    Mf   = (float*)smem;                                   // 64 x 520 f32
    unsigned* Mb   = (unsigned*)(smem + 133120);                     // 64 x 257 dw (bf16 pairs)
    unsigned* Xb   = (unsigned*)(smem + 133120 + 65792);             // 64 x 257 dw
    float*    outS = (float*)(smem + 133120 + 2 * 65792);            // 64 x 64 f32 (contiguous: TDM src)
    float*    ewS  = outS + 64 * 64;
    float*    awS  = ewS + DDIM;
    float*    gwS  = awS + DDIM;
    float*    kmS  = gwS + DDIM;                                     // k_mean [512]
    float*    etaS = kmS + DDIM;                                     // per-token [64]
    float*    alpS = etaS + 64;
    float*    invnS= alpS + 64;
    float*    MkS  = invnS + 64;
    float*    errS = MkS + 64;
    float*    red  = errS + 64;                                      // [256]
    float*    scal = red + 256;                                      // [0]=eta [1]=alpha [2]=gate_dot

    const int tid  = threadIdx.x;
    const int w    = tid >> 5, lane = tid & 31;
    const int bid  = blockIdx.x;
    const int b    = bid >> 3;       // batch
    const int r    = bid & 7;        // row block: rows r*64 .. r*64+63
    const int t    = tid >> 2;       // token / local-row owner (0..63)
    const int q    = tid & 3;        // quarter of a 512-row

    const float eta_b = eta_bp[0], alpha_b = alpha_bp[0], gate_b = gate_bp[0];

    // ---- stage projection vectors to LDS
    for (int i = tid; i < DDIM; i += 256) { ewS[i] = eta_w[i]; awS[i] = alpha_w[i]; gwS[i] = gate_w[i]; }

    // ---- load M_init rows [r*64, r*64+64) into padded fp32 LDS (pitch 520)
#if USE_TDM
    if (w == 0) {
        v4u g0; v8i_t g1;
        // pad: every 256 dwords insert 4 dwords  -> row pitch 520 dwords
        tdm_desc(lds_off(Mf),
                 (unsigned long long)(size_t)(Minit + (size_t)(r * ROWS) * DDIM),
                 512u, 64u, 512u, 64u, 512ull, /*pad*/1, /*interval=256dw*/7, /*amount=4dw*/3,
                 g0, g1);
        v4i_t z4 = {0, 0, 0, 0};
        v8i_t z8 = {0, 0, 0, 0, 0, 0, 0, 0};
        __builtin_amdgcn_tensor_load_to_lds(g0, g1, z4, z4, z8, 0);
    }
    WAIT_TENSOR();
    __syncthreads();
#else
    for (int j = 0; j < 32; ++j) {
        int k = q * 4 + 16 * j;
        int idx = t * MF_PITCH + k + ((k >= 256) ? 4 : 0);
        *(float4*)(Mf + idx) = *(const float4*)(Minit + (size_t)(r * ROWS + t) * DDIM + k);
    }
    __syncthreads();
#endif

    // ---- bf16 mirror of M for the WMMA pipe
    for (int j = 0; j < 32; ++j) {
        int k = q * 4 + 16 * j;
        int idx = t * MF_PITCH + k + ((k >= 256) ? 4 : 0);
        float4 m = *(const float4*)(Mf + idx);
        Mb[t * BF_PITCH + (k >> 1)]     = pk_bf16(m.x, m.y);
        Mb[t * BF_PITCH + (k >> 1) + 1] = pk_bf16(m.z, m.w);
    }
    __syncthreads();

    // =================== sequential chunk scan ===================
    for (int ch = 0; ch < NCHUNK; ++ch) {
        WAIT_TENSOR();          // previous chunk's out tile TDM store done (wave0)
        __syncthreads();        // before anyone rewrites Xb/outS

        // ---- stage chunk tokens: global -> fp32 stats (regs) + bf16 LDS
        {
            const float* xrow = x + (((size_t)b * 4096) + (size_t)ch * CHUNK_T + t) * DDIM;
            float se = 0.f, sa = 0.f, sn = 0.f;
            for (int j = 0; j < 32; ++j) {
                int k = q * 4 + 16 * j;
                float4 v = *(const float4*)(xrow + k);
                se += v.x * ewS[k] + v.y * ewS[k + 1] + v.z * ewS[k + 2] + v.w * ewS[k + 3];
                sa += v.x * awS[k] + v.y * awS[k + 1] + v.z * awS[k + 2] + v.w * awS[k + 3];
                sn += v.x * v.x + v.y * v.y + v.z * v.z + v.w * v.w;
                Xb[t * BF_PITCH + (k >> 1)]     = pk_bf16(v.x, v.y);
                Xb[t * BF_PITCH + (k >> 1) + 1] = pk_bf16(v.z, v.w);
            }
            se += __shfl_xor(se, 1); se += __shfl_xor(se, 2);
            sa += __shfl_xor(sa, 1); sa += __shfl_xor(sa, 2);
            sn += __shfl_xor(sn, 1); sn += __shfl_xor(sn, 2);
            if (q == 0) {
                etaS[t]  = sigm(se + eta_b) * MAX_LR;
                alpS[t]  = MIN_DEC + sigm(sa + alpha_b) * (1.0f - MIN_DEC);
                invnS[t] = 1.0f / fmaxf(sqrtf(sn), 1e-5f);
            }
        }
        __syncthreads();

        // ---- k_mean (thread owns dims 2*tid, 2*tid+1) and eta/alpha chunk means
        {
            float km0 = 0.f, km1 = 0.f;
            for (int tt = 0; tt < 64; ++tt) {
                unsigned u = Xb[tt * BF_PITCH + tid];
                float inw = invnS[tt];
                km0 += bf_lo(u) * inw;
                km1 += bf_hi(u) * inw;
            }
            kmS[2 * tid]     = km0 * (1.0f / 64.0f);
            kmS[2 * tid + 1] = km1 * (1.0f / 64.0f);
            if (w == 0) {
                float s = etaS[lane] + etaS[lane + 32];
                s += __shfl_xor(s, 16); s += __shfl_xor(s, 8); s += __shfl_xor(s, 4);
                s += __shfl_xor(s, 2);  s += __shfl_xor(s, 1);
                if (lane == 0) scal[0] = s * (1.0f / 64.0f);
            } else if (w == 1) {
                float s = alpS[lane] + alpS[lane + 32];
                s += __shfl_xor(s, 16); s += __shfl_xor(s, 8); s += __shfl_xor(s, 4);
                s += __shfl_xor(s, 2);  s += __shfl_xor(s, 1);
                if (lane == 0) scal[1] = s * (1.0f / 64.0f);
            }
        }
        __syncthreads();

        // ---- gate-dot partials and Mk = M(row) . k_mean
        red[tid] = kmS[2 * tid] * gwS[2 * tid] + kmS[2 * tid + 1] * gwS[2 * tid + 1];
        {
            float mk = 0.f;
            for (int j = 0; j < 32; ++j) {
                int k = q * 4 + 16 * j;
                int idx = t * MF_PITCH + k + ((k >= 256) ? 4 : 0);
                float4 m  = *(const float4*)(Mf + idx);
                float4 km = *(const float4*)(kmS + k);
                mk += m.x * km.x + m.y * km.y + m.z * km.z + m.w * km.w;
            }
            mk += __shfl_xor(mk, 1); mk += __shfl_xor(mk, 2);
            if (q == 0) MkS[t] = mk;
        }
        __syncthreads();
        if (tid < 32) {
            float s = 0.f;
            for (int i = 0; i < 8; ++i) s += red[tid + 32 * i];
            s += __shfl_xor(s, 16); s += __shfl_xor(s, 8); s += __shfl_xor(s, 4);
            s += __shfl_xor(s, 2);  s += __shfl_xor(s, 1);
            if (tid == 0) scal[2] = s;
        }
        __syncthreads();

        // ---- retrieval GEMM: outS[64 tok x 64 row] = X(bf16) @ M(bf16)^T, fp32 acc
        {
            const int mt = w & 3, ntb = (w >> 2) << 1;   // wave: 1 token-tile x 2 row-tiles
            const int la = lane & 15, lh = lane >> 4;
            const unsigned* Xu  = Xb + (mt * 16 + la) * BF_PITCH;
            const unsigned* B0u = Mb + (ntb * 16 + la) * BF_PITCH;
            const unsigned* B1u = B0u + 16 * BF_PITCH;
            v8f c0 = {0.f,0.f,0.f,0.f,0.f,0.f,0.f,0.f};
            v8f c1 = {0.f,0.f,0.f,0.f,0.f,0.f,0.f,0.f};
            for (int kb = 0; kb < DDIM; kb += 32) {
                const int base = kb >> 1;
                union { unsigned u[8]; v16bf v; } A, B0, B1;
#pragma unroll
                for (int j = 0; j < 8; ++j) {            // A: lanes<16 K 0..15
                    A.u[j] = Xu[base + lh * 8 + j];
                }
#pragma unroll
                for (int j = 0; j < 8; ++j) {            // B: lanes<16 K 0..15
                    B0.u[j] = B0u[base + lh * 8 + j];
                    B1.u[j] = B1u[base + lh * 8 + j];
                }
                c0 = __builtin_amdgcn_wmma_f32_16x16x32_bf16(false, A.v, false, B0.v, (short)0, c0, false, false);
                c1 = __builtin_amdgcn_wmma_f32_16x16x32_bf16(false, A.v, false, B1.v, (short)0, c1, false, false);
            }
#pragma unroll
            for (int vv = 0; vv < 8; ++vv) {
                int tt = mt * 16 + vv + lh * 8;
                outS[tt * 64 + ntb * 16 + la]       = c0[vv];
                outS[tt * 64 + (ntb + 1) * 16 + la] = c1[vv];
            }
        }
        __syncthreads();

        // ---- v_target column means and err
        if (tid < 64) {
            float s = 0.f;
            for (int tt = 0; tt < 64; ++tt) s += outS[tt * 64 + tid];
            errS[tid] = s * (1.0f / 64.0f) - MkS[tid];
        }
        __syncthreads();

        // ---- async store of out tile via TDM (overlaps with the M update below)
#if USE_TDM
        if (w == 0) {
            unsigned long long ga = (unsigned long long)(size_t)
                (out + (((size_t)b * 4096 + (size_t)ch * CHUNK_T) * DDIM + (size_t)r * ROWS));
            v4u g0; v8i_t g1;
            tdm_desc(lds_off(outS), ga, 64u, 64u, 64u, 64u, 512ull, 0, 0, 0, g0, g1);
            v4i_t z4 = {0, 0, 0, 0};
            v8i_t z8 = {0, 0, 0, 0, 0, 0, 0, 0};
            __builtin_amdgcn_tensor_store_from_lds(g0, g1, z4, z4, z8, 0);
        }
#else
        {
            float* orow = out + (((size_t)b * 4096 + (size_t)ch * CHUNK_T + t) * DDIM + (size_t)r * ROWS);
            for (int j = 0; j < 4; ++j)
                *(float4*)(orow + q * 16 + 4 * j) = *(const float4*)(outS + t * 64 + q * 16 + 4 * j);
        }
#endif

        // ---- delta update: M = c1*M + c2*err[o]*k_mean, accumulate ||M||^2 partial
        const float etaM  = scal[0];
        const float alphM = scal[1];
        const float gate  = sigm(scal[2] + gate_b);
        const float c1s   = gate * alphM + (1.0f - gate);
        const float c2s   = gate * etaM;
        float sq = 0.f;
        {
            const float e = errS[t] * c2s;
            for (int j = 0; j < 32; ++j) {
                int k = q * 4 + 16 * j;
                int idx = t * MF_PITCH + k + ((k >= 256) ? 4 : 0);
                float4 m  = *(const float4*)(Mf + idx);
                float4 km = *(const float4*)(kmS + k);
                m.x = c1s * m.x + e * km.x;
                m.y = c1s * m.y + e * km.y;
                m.z = c1s * m.z + e * km.z;
                m.w = c1s * m.w + e * km.w;
                sq += m.x * m.x + m.y * m.y + m.z * m.z + m.w * m.w;
                *(float4*)(Mf + idx) = m;
            }
        }
        red[tid] = sq;
        __syncthreads();
        if (tid < 32) {
            float s = 0.f;
            for (int i = 0; i < 8; ++i) s += red[tid + 32 * i];
            s += __shfl_xor(s, 16); s += __shfl_xor(s, 8); s += __shfl_xor(s, 4);
            s += __shfl_xor(s, 2);  s += __shfl_xor(s, 1);
            if (tid == 0) {
                partials[(b * 64 + ch) * 8 + r] = s;
                __hip_atomic_fetch_add(&arrive[b * 64 + ch], 1,
                                       __ATOMIC_RELEASE, __HIP_MEMORY_SCOPE_AGENT);
            }
        }
        // all threads spin on the per-chunk arrival counter (acquire)
        while (__hip_atomic_load(&arrive[b * 64 + ch],
                                 __ATOMIC_ACQUIRE, __HIP_MEMORY_SCOPE_AGENT) < 8)
            __builtin_amdgcn_s_sleep(2);

        // deterministic fixed-order sum of the 8 partials
        float fro2 = 0.f;
        {
            const float* pp = partials + (b * 64 + ch) * 8;
#pragma unroll
            for (int i = 0; i < 8; ++i) fro2 += pp[i];
        }
        const float scale = fminf(MAX_NRM / (sqrtf(fro2) + 1e-6f), 1.0f);

        // ---- apply clamp scale; refresh bf16 mirror for next chunk's GEMM
        for (int j = 0; j < 32; ++j) {
            int k = q * 4 + 16 * j;
            int idx = t * MF_PITCH + k + ((k >= 256) ? 4 : 0);
            float4 m = *(const float4*)(Mf + idx);
            m.x *= scale; m.y *= scale; m.z *= scale; m.w *= scale;
            *(float4*)(Mf + idx) = m;
            Mb[t * BF_PITCH + (k >> 1)]     = pk_bf16(m.x, m.y);
            Mb[t * BF_PITCH + (k >> 1) + 1] = pk_bf16(m.z, m.w);
        }
        __syncthreads();
    }

    // ---- final M state -> d_out tail region [16,512,512]
    {
        float* mrow = Mfin + (((size_t)b * DDIM) + (size_t)r * ROWS + t) * DDIM;
        for (int j = 0; j < 32; ++j) {
            int k = q * 4 + 16 * j;
            int idx = t * MF_PITCH + k + ((k >= 256) ? 4 : 0);
            *(float4*)(mrow + k) = *(const float4*)(Mf + idx);
        }
    }
}

extern "C" void kernel_launch(void* const* d_in, const int* in_sizes, int n_in,
                              void* d_out, int out_size, void* d_ws, size_t ws_size,
                              hipStream_t stream) {
    const float* x       = (const float*)d_in[0];
    const float* Minit   = (const float*)d_in[1];
    const float* eta_w   = (const float*)d_in[2];
    const float* eta_b   = (const float*)d_in[3];
    const float* alpha_w = (const float*)d_in[4];
    const float* alpha_b = (const float*)d_in[5];
    const float* gate_w  = (const float*)d_in[6];
    const float* gate_b  = (const float*)d_in[7];

    float* out  = (float*)d_out;
    float* Mfin = out + (size_t)NBATCH * 4096 * DDIM;   // tuple: (out, M_final) concatenated

    int*   arrive   = (int*)d_ws;                        // [16*64] counters
    float* partials = (float*)((char*)d_ws + 4096);      // [16*64*8] fro^2 partials

    // counters must be zero every call (d_ws is not re-poisoned between replays)
    (void)hipMemsetAsync(d_ws, 0, 4096, stream);

    dim3 grid(NBATCH * RBLK), block(256);
    size_t shmem = 291648;   // Mf32(133120) + Mbf(65792) + Xbf(65792) + out(16384) + scalars(10560)
    hipLaunchKernelGGL(delta_mem_kernel, grid, block, shmem, stream,
                       x, Minit, eta_w, eta_b, alpha_w, alpha_b, gate_w, gate_b,
                       out, Mfin, arrive, partials);
}
